// CodeNNModelUML_65274912964688
// MI455X (gfx1250) — matrix operand order repaired and checked by reading, not compiled, not run on hardware
//
#include <hip/hip_runtime.h>

// ---------------------------------------------------------------------------
// Problem constants (from reference): B=256, L=200, D=512, Vs=20000, 15 steps
// ---------------------------------------------------------------------------
#define B_     256
#define L_     200
#define D_     512
#define VS_    20000
#define NPAD_  20224   // 79 * 256 : projection N padded to block-tile multiple
#define STEPS_ 15

typedef __attribute__((ext_vector_type(16))) __bf16 v16bf;
typedef __attribute__((ext_vector_type(8)))  __bf16 v8bf;
typedef __attribute__((ext_vector_type(8)))  float  v8f;

// ----------------------------- bf16 helpers --------------------------------
__device__ __forceinline__ float bfu2f(unsigned short h) {
    unsigned u = ((unsigned)h) << 16;
    return __builtin_bit_cast(float, u);
}
__device__ __forceinline__ unsigned short f2bfu(float f) {
    unsigned u = __builtin_bit_cast(unsigned, f);
    unsigned r = u + 0x7FFFu + ((u >> 16) & 1u);   // round-to-nearest-even
    return (unsigned short)(r >> 16);
}
__device__ __forceinline__ float sigmoidf_(float x) {
    return 1.0f / (1.0f + expf(-x));
}

// --------------------- one-time-per-call prep kernels ----------------------
__global__ void convert_f32_to_bf16(const float* __restrict__ in,
                                    unsigned short* __restrict__ out,
                                    long long n) {
    for (long long i = (long long)blockIdx.x * 256 + threadIdx.x; i < n;
         i += (long long)gridDim.x * 256)
        out[i] = f2bfu(in[i]);
}

// converts first nSrc elements, zero-fills up to nTot (weight row padding)
__global__ void convert_f32_to_bf16_pad(const float* __restrict__ in,
                                        unsigned short* __restrict__ out,
                                        long long nSrc, long long nTot) {
    for (long long i = (long long)blockIdx.x * 256 + threadIdx.x; i < nTot;
         i += (long long)gridDim.x * 256)
        out[i] = (i < nSrc) ? f2bfu(in[i]) : (unsigned short)0;
}

__global__ void gather_code_bf16(const int* __restrict__ mc,
                                 const float* __restrict__ tbl,
                                 unsigned short* __restrict__ out) {
    const size_t total = (size_t)B_ * L_ * D_;
    for (size_t i = (size_t)blockIdx.x * 256 + threadIdx.x; i < total;
         i += (size_t)gridDim.x * 256) {
        int    d  = (int)(i & (D_ - 1));
        size_t bl = i >> 9;                       // D_ == 512
        int tok   = mc[bl];
        out[i] = f2bfu(tbl[(size_t)tok * D_ + d]);
    }
}

__global__ void init_state_kernel(float* h, float* c, unsigned short* hb, int* prev) {
    int i = blockIdx.x * 256 + threadIdx.x;
    if (i < B_ * D_) { h[i] = 0.0f; c[i] = 0.0f; hb[i] = 0; }
    if (i < B_) prev[i] = 1;                      // SOS token
}

// --------------------------- per-step kernels ------------------------------
__global__ void gather_x_kernel(const int* __restrict__ prev,
                                const float* __restrict__ tbl,
                                unsigned short* __restrict__ x) {
    int i = blockIdx.x * 256 + threadIdx.x;       // i < B_*D_
    int b = i >> 9, d = i & (D_ - 1);
    x[i] = f2bfu(tbl[(size_t)prev[b] * D_ + d]);
}

__global__ void lstm_update_kernel(const float* __restrict__ gates,
                                   float* __restrict__ h, float* __restrict__ c,
                                   unsigned short* __restrict__ hb) {
    int i = blockIdx.x * 256 + threadIdx.x;       // i < B_*D_
    int b = i >> 9, d = i & (D_ - 1);
    const float* g = gates + (size_t)b * (4 * D_);
    float ig = sigmoidf_(g[d]);
    float fg = sigmoidf_(g[D_ + d]);
    float gg = tanhf(g[2 * D_ + d]);
    float og = sigmoidf_(g[3 * D_ + d]);
    float cn = fg * c[i] + ig * gg;
    float hn = og * tanhf(cn);
    c[i] = cn; h[i] = hn; hb[i] = f2bfu(hn);
}

// one block per batch row: scores -> softmax -> context, all in one pass
__global__ __launch_bounds__(256)
void attention_kernel(const float* __restrict__ h,
                      const unsigned short* __restrict__ code,   // [B,L,D] bf16
                      unsigned short* __restrict__ tOut) {       // [B,D] bf16
    __shared__ float hs[D_];
    __shared__ float sc[256];
    __shared__ float red[256];
    const int b = blockIdx.x, tid = threadIdx.x;

    for (int i = tid; i < D_; i += 256) hs[i] = h[(size_t)b * D_ + i];
    __syncthreads();

    float v = -3.4e38f;
    if (tid < L_) {
        const unsigned short* crow = code + ((size_t)b * L_ + tid) * D_;
        float a = 0.0f;
        for (int d = 0; d < D_; ++d) a += hs[d] * bfu2f(crow[d]);
        v = a;
    }
    red[tid] = v;
    __syncthreads();
    for (int s = 128; s > 0; s >>= 1) {
        if (tid < s) red[tid] = fmaxf(red[tid], red[tid + s]);
        __syncthreads();
    }
    const float m = red[0];
    __syncthreads();
    float e = (tid < L_) ? expf(v - m) : 0.0f;
    red[tid] = e;
    __syncthreads();
    for (int s = 128; s > 0; s >>= 1) {
        if (tid < s) red[tid] += red[tid + s];
        __syncthreads();
    }
    const float inv = 1.0f / red[0];
    __syncthreads();
    sc[tid] = e * inv;
    __syncthreads();

    for (int d = tid; d < D_; d += 256) {
        float a = 0.0f;
        for (int l = 0; l < L_; ++l)
            a += sc[l] * bfu2f(code[((size_t)b * L_ + l) * D_ + d]);
        tOut[(size_t)b * D_ + d] = f2bfu(a);
    }
}

__global__ __launch_bounds__(256)
void argmax_kernel(const float* __restrict__ logits, long long rowStride,
                   int* __restrict__ prev) {
    const int b = blockIdx.x, tid = threadIdx.x;
    const float* row = logits + (size_t)b * rowStride;
    float bv = -3.4e38f; int bi = 0x7fffffff;
    for (int i = tid; i < VS_; i += 256) {
        float x = row[i];
        if (x > bv) { bv = x; bi = i; }
    }
    __shared__ float rv[256];
    __shared__ int   ri[256];
    rv[tid] = bv; ri[tid] = bi;
    __syncthreads();
    for (int s = 128; s > 0; s >>= 1) {
        if (tid < s) {
            if (rv[tid + s] > rv[tid] ||
                (rv[tid + s] == rv[tid] && ri[tid + s] < ri[tid])) {
                rv[tid] = rv[tid + s]; ri[tid] = ri[tid + s];
            }
        }
        __syncthreads();
    }
    if (tid == 0) prev[b] = ri[0];
}

// ---------------------------------------------------------------------------
// Dual-source bf16 WMMA GEMM:  Out[M,N] = A1@W1^T (+ A2@W2^T) + bias (opt tanh)
//   A*: [M, 512] bf16 row-major     W*: [>= grid.x*256, 512] bf16 row-major
// Block: 256 thr = 8 waves as 2(M) x 4(N); block tile 64x256.
// Wave tile 32x64: 2 A fragments x 4 weight fragments -> 8 accumulators,
// so each weight fragment feeds 2 WMMAs (halves L2 weight traffic vs 16xN).
// All WMMAs are unconditional (weights padded); only stores check Nstore.
// A panels are staged memory->LDS with gfx1250 async DMA loads (ASYNCcnt).
// LDS row stride padded to 520 ushorts to avoid 16-way bank conflicts.
// ---------------------------------------------------------------------------
#define SA_STRIDE 520   // 512 + 8 (one uint4 of pad)
#define MBLK      64    // rows per block (4 blocks cover B=256)

__global__ __launch_bounds__(256)
void gemm_dual_bf16_wmma(const unsigned short* __restrict__ A1,
                         const unsigned short* __restrict__ W1,
                         const float* __restrict__ bias1,
                         const unsigned short* __restrict__ A2,
                         const unsigned short* __restrict__ W2,
                         const float* __restrict__ bias2,
                         float* __restrict__ outF, long long ldF,
                         unsigned short* __restrict__ outB, long long ldB,
                         int Nstore, int doTanh) {
    __shared__ unsigned short sA1[MBLK * SA_STRIDE];
    __shared__ unsigned short sA2[MBLK * SA_STRIDE];

    const int tid  = threadIdx.x;
    const int lane = tid & 31;
    const int wave = tid >> 5;
    const int wm   = wave >> 2;                 // 0..1 : 32-row M sub-tile
    const int wn   = wave & 3;                  // 0..3 : 64-col N sub-tile
    const int mBlock = blockIdx.y * MBLK;
    const int n0     = blockIdx.x * 256 + wn * 64;
    const bool hasA2 = (A2 != nullptr);

    // ---- async DMA staging of A panels into LDS (ASYNCcnt-tracked) ----
    // Generic LDS pointers: addr[31:0] IS the LDS address (aperture mapping),
    // which is what global_load_async_to_lds_* expects in its VDST VGPR.
    {
        const unsigned short* g1 = A1 + (size_t)mBlock * D_;
        const int r = tid >> 2, c0 = (tid & 3) * 16;    // 64 rows x 64 uint4
        unsigned lds1 = (unsigned)(size_t)&sA1[0] + (unsigned)((r * 65) * 16);
#pragma unroll
        for (int c = 0; c < 16; ++c) {
            unsigned ldsAddr = lds1 + (unsigned)((c0 + c) * 16);
            unsigned goff    = (unsigned)((r * 64 + c0 + c) * 16);
            asm volatile("global_load_async_to_lds_b128 %0, %1, %2"
                         :: "v"(ldsAddr), "v"(goff), "s"(g1) : "memory");
        }
        if (hasA2) {
            const unsigned short* g2 = A2 + (size_t)mBlock * D_;
            unsigned lds2 = (unsigned)(size_t)&sA2[0] + (unsigned)((r * 65) * 16);
#pragma unroll
            for (int c = 0; c < 16; ++c) {
                unsigned ldsAddr = lds2 + (unsigned)((c0 + c) * 16);
                unsigned goff    = (unsigned)((r * 64 + c0 + c) * 16);
                asm volatile("global_load_async_to_lds_b128 %0, %1, %2"
                             :: "v"(ldsAddr), "v"(goff), "s"(g2) : "memory");
            }
        }
        asm volatile("s_wait_asynccnt 0x0" ::: "memory");
    }
    __syncthreads();

    const int l15   = lane & 15;
    const int hi    = lane >> 4;                // 0 or 1 (half-wave)
    const int rowL0 = wm * 32 + l15;            // A rows held by this lane
    const int rowL1 = rowL0 + 16;

    v8f acc[2][4] = {};

#pragma unroll 4
    for (int k0 = 0; k0 < D_; k0 += 32) {
        // 16-bit A fragments (matching B): half-wave 0 -> K {0..15},
        // half-wave 1 -> K {16..31} relative to k0.
        const int ka = k0 + hi * 16;
        v8bf a0lo = *reinterpret_cast<const v8bf*>(&sA1[rowL0 * SA_STRIDE + ka]);
        v8bf a0hi = *reinterpret_cast<const v8bf*>(&sA1[rowL0 * SA_STRIDE + ka + 8]);
        v8bf a1lo = *reinterpret_cast<const v8bf*>(&sA1[rowL1 * SA_STRIDE + ka]);
        v8bf a1hi = *reinterpret_cast<const v8bf*>(&sA1[rowL1 * SA_STRIDE + ka + 8]);
        v16bf fa0 = __builtin_shufflevector(a0lo, a0hi, 0, 1, 2, 3, 4, 5, 6, 7,
                                            8, 9, 10, 11, 12, 13, 14, 15);
        v16bf fa1 = __builtin_shufflevector(a1lo, a1hi, 0, 1, 2, 3, 4, 5, 6, 7,
                                            8, 9, 10, 11, 12, 13, 14, 15);
        // 16-bit B fragment: lane = column, half-wave 1 holds K+16..31 -> one
        // contiguous 32B chunk of the [N,K] row-major weight row.
        const int kb = k0 + hi * 16;
#pragma unroll
        for (int j = 0; j < 4; ++j) {
            const v16bf w = *reinterpret_cast<const v16bf*>(
                &W1[(size_t)(n0 + j * 16 + l15) * D_ + kb]);
            acc[0][j] = __builtin_amdgcn_wmma_f32_16x16x32_bf16(
                false, fa0, false, w, (short)0, acc[0][j], false, false);
            acc[1][j] = __builtin_amdgcn_wmma_f32_16x16x32_bf16(
                false, fa1, false, w, (short)0, acc[1][j], false, false);
        }
        if (hasA2) {
            v8bf b0lo = *reinterpret_cast<const v8bf*>(&sA2[rowL0 * SA_STRIDE + ka]);
            v8bf b0hi = *reinterpret_cast<const v8bf*>(&sA2[rowL0 * SA_STRIDE + ka + 8]);
            v8bf b1lo = *reinterpret_cast<const v8bf*>(&sA2[rowL1 * SA_STRIDE + ka]);
            v8bf b1hi = *reinterpret_cast<const v8bf*>(&sA2[rowL1 * SA_STRIDE + ka + 8]);
            v16bf fb0 = __builtin_shufflevector(b0lo, b0hi, 0, 1, 2, 3, 4, 5, 6, 7,
                                                8, 9, 10, 11, 12, 13, 14, 15);
            v16bf fb1 = __builtin_shufflevector(b1lo, b1hi, 0, 1, 2, 3, 4, 5, 6, 7,
                                                8, 9, 10, 11, 12, 13, 14, 15);
#pragma unroll
            for (int j = 0; j < 4; ++j) {
                const v16bf w = *reinterpret_cast<const v16bf*>(
                    &W2[(size_t)(n0 + j * 16 + l15) * D_ + kb]);
                acc[0][j] = __builtin_amdgcn_wmma_f32_16x16x32_bf16(
                    false, fb0, false, w, (short)0, acc[0][j], false, false);
                acc[1][j] = __builtin_amdgcn_wmma_f32_16x16x32_bf16(
                    false, fb1, false, w, (short)0, acc[1][j], false, false);
            }
        }
        // gfx1250 global_prefetch_b8, near-cache scope, for the next K panel
        if (k0 + 32 < D_)
            __builtin_prefetch(&W1[(size_t)(n0 + l15) * D_ + kb + 32], 0, 3);
    }

    // Epilogue. C/D layout: VGPR e holds row e (lanes 0-15) / row e+8 (16-31).
#pragma unroll
    for (int j = 0; j < 4; ++j) {
        const int col = n0 + j * 16 + l15;
        if (col < Nstore) {
            float bias = bias1[col];
            if (hasA2) bias += bias2[col];
#pragma unroll
            for (int g = 0; g < 2; ++g) {
                const int rowBase = mBlock + wm * 32 + g * 16 + hi * 8;
#pragma unroll
                for (int e = 0; e < 8; ++e) {
                    float v = acc[g][j][e] + bias;
                    if (doTanh) v = tanhf(v);
                    const int row = rowBase + e;
                    if (outF) outF[(size_t)row * ldF + col] = v;
                    if (outB) outB[(size_t)row * ldB + col] = f2bfu(v);
                }
            }
        }
    }
}

// ---------------------------------------------------------------------------
// Host driver: 15 sequential decode steps, all on `stream`
// ---------------------------------------------------------------------------
extern "C" void kernel_launch(void* const* d_in, const int* in_sizes, int n_in,
                              void* d_out, int out_size, void* d_ws, size_t ws_size,
                              hipStream_t stream) {
    (void)in_sizes; (void)n_in; (void)out_size; (void)ws_size;

    const int*   method_code = (const int*)  d_in[0];
    const float* code_tbl    = (const float*)d_in[1];
    const float* sum_tbl     = (const float*)d_in[2];
    const float* w_ih        = (const float*)d_in[3];
    const float* w_hh        = (const float*)d_in[4];
    const float* b_ih        = (const float*)d_in[5];
    const float* b_hh        = (const float*)d_in[6];
    const float* t_w         = (const float*)d_in[7];
    const float* t_b         = (const float*)d_in[8];
    const float* h_w         = (const float*)d_in[9];
    const float* h_b         = (const float*)d_in[10];
    const float* p_w         = (const float*)d_in[11];
    const float* p_b         = (const float*)d_in[12];
    float* out = (float*)d_out;

    char*  ws  = (char*)d_ws;
    size_t off = 0;
    auto take = [&](size_t bytes) -> char* {
        char* p = ws + off;
        off += (bytes + 255) & ~(size_t)255;
        return p;
    };
    unsigned short* codeB = (unsigned short*)take((size_t)B_ * L_ * D_ * 2);
    unsigned short* pwB   = (unsigned short*)take((size_t)NPAD_ * D_ * 2);   // padded
    unsigned short* wihB  = (unsigned short*)take((size_t)4 * D_ * D_ * 2);
    unsigned short* whhB  = (unsigned short*)take((size_t)4 * D_ * D_ * 2);
    unsigned short* twB   = (unsigned short*)take((size_t)D_ * D_ * 2);
    unsigned short* hwB   = (unsigned short*)take((size_t)D_ * D_ * 2);
    unsigned short* xB    = (unsigned short*)take((size_t)B_ * D_ * 2);
    unsigned short* hB    = (unsigned short*)take((size_t)B_ * D_ * 2);
    unsigned short* tB    = (unsigned short*)take((size_t)B_ * D_ * 2);
    unsigned short* tpB   = (unsigned short*)take((size_t)B_ * D_ * 2);
    float* hF     = (float*)take((size_t)B_ * D_ * 4);
    float* cF     = (float*)take((size_t)B_ * D_ * 4);
    float* gatesF = (float*)take((size_t)B_ * 4 * D_ * 4);
    int*   prev   = (int*)take((size_t)B_ * 4);

    const dim3 blk(256);

    // One-time (per call) precision conversion + embedding gather
    convert_f32_to_bf16_pad<<<2048, blk, 0, stream>>>(
        p_w, pwB, (long long)VS_ * D_, (long long)NPAD_ * D_);
    convert_f32_to_bf16<<<1024, blk, 0, stream>>>(w_ih, wihB, (long long)4 * D_ * D_);
    convert_f32_to_bf16<<<1024, blk, 0, stream>>>(w_hh, whhB, (long long)4 * D_ * D_);
    convert_f32_to_bf16<<<256,  blk, 0, stream>>>(t_w, twB, (long long)D_ * D_);
    convert_f32_to_bf16<<<256,  blk, 0, stream>>>(h_w, hwB, (long long)D_ * D_);
    gather_code_bf16<<<4096, blk, 0, stream>>>(method_code, code_tbl, codeB);
    init_state_kernel<<<512, blk, 0, stream>>>(hF, cF, hB, prev);

    for (int s = 0; s < STEPS_; ++s) {
        gather_x_kernel<<<512, blk, 0, stream>>>(prev, sum_tbl, xB);
        // gates = x@w_ih^T + h_{t-1}@w_hh^T + b_ih + b_hh   (N = 2048)
        gemm_dual_bf16_wmma<<<dim3(8, 4), blk, 0, stream>>>(
            xB, wihB, b_ih, hB, whhB, b_hh,
            gatesF, (long long)(4 * D_), nullptr, 0, 4 * D_, 0);
        lstm_update_kernel<<<512, blk, 0, stream>>>(gatesF, hF, cF, hB);
        attention_kernel<<<B_, blk, 0, stream>>>(hF, codeB, tB);
        // tanh(t@t_w^T + h_t@h_w^T + t_b + h_b) -> bf16     (N = 512)
        gemm_dual_bf16_wmma<<<dim3(2, 4), blk, 0, stream>>>(
            tB, twB, t_b, hB, hwB, h_b,
            nullptr, 0, tpB, (long long)D_, D_, 1);
        // logits = tanh_pre @ p_w^T + p_b -> d_out[:, s, :] (N = 20000, pad 20224)
        gemm_dual_bf16_wmma<<<dim3(79, 4), blk, 0, stream>>>(
            tpB, pwB, p_b, nullptr, nullptr, nullptr,
            out + (size_t)s * VS_, (long long)STEPS_ * VS_, nullptr, 0, VS_, 0);
        argmax_kernel<<<B_, blk, 0, stream>>>(
            out + (size_t)s * VS_, (long long)STEPS_ * VS_, prev);
    }
}
